// CLRerHead_53730040873749
// MI455X (gfx1250) — matrix operand order repaired and checked by reading, not compiled, not run on hardware
//
#include <hip/hip_runtime.h>

#define PI_F 3.14159265358979323846f

// ---------------------------------------------------------------------------
// Problem constants
// ---------------------------------------------------------------------------
#define Bn    32
#define Cn    64
#define NPp   192
#define NSn   36
#define NOn   72
#define MIDn  48
#define MROWS (Bn * NPp)      // 6144
#define SPAD  44              // 36 + 4 pad each side
#define KFC   (Cn * NSn)      // 2304
#define KC    64              // K-panel for LDS-staged fc GEMM

typedef _Float16 h8  __attribute__((ext_vector_type(8)));
typedef _Float16 h16 __attribute__((ext_vector_type(16)));
typedef float    f8  __attribute__((ext_vector_type(8)));

#define WMMA_F32_F16(a, b, c) \
  __builtin_amdgcn_wmma_f32_16x16x32_f16(false, (a), false, (b), (short)0, (c), false, false)

// ---------------------------------------------------------------------------
// CDNA5 async global->LDS DMA (CDNA5 ISA, VGLOBAL op 98).
// VDST = LDS byte address (low 32 bits of the flat shared pointer, per ISA
// §10.2 LDS aperture). Tracked by ASYNCcnt.
// ---------------------------------------------------------------------------
__device__ __forceinline__ void async_ld_b128(uint32_t lds_off, const _Float16* g) {
  asm volatile("global_load_async_to_lds_b128 %0, %1, off"
               :: "v"(lds_off), "v"(g)
               : "memory");
}
__device__ __forceinline__ void wait_async0() {
  asm volatile("s_wait_asynccnt 0x0" ::: "memory");
}

// ---------------------------------------------------------------------------
// Fragment loaders per CDNA5 wave32 WMMA VGPR layouts (CDNA5 ISA)
// A (16x32 f16, row-major): lane l -> row m=l&15, K = {kb..kb+7, kb+16..kb+23},
//   kb = 8*(l>>4).  Two 16-byte loads.
// B given as B^T (N x K row-major): lane l -> col n=l&15, K = {kb..kb+7, kb+16..kb+23},
//   kb = 8*(l>>4).  Two 16-byte loads.
// Works for both global and LDS-resident (ds_load_b128) operands.
// ---------------------------------------------------------------------------
__device__ __forceinline__ h16 frag_a16(const _Float16* base, int lda, int lane) {
  int m  = lane & 15;
  int kb = (lane >> 4) << 3;
  const _Float16* p = base + (size_t)m * lda + kb;
  h8 lo = *(const h8*)(p);
  h8 hi = *(const h8*)(p + 16);
  h16 r;
#pragma unroll
  for (int i = 0; i < 8; ++i) { r[i] = lo[i]; r[i + 8] = hi[i]; }
  return r;
}

__device__ __forceinline__ h16 frag_b16(const _Float16* base, int ldb, int lane) {
  int n  = lane & 15;
  int kb = (lane >> 4) << 3;
  const _Float16* p = base + (size_t)n * ldb + kb;
  h8 lo = *(const h8*)(p);
  h8 hi = *(const h8*)(p + 16);
  h16 r;
#pragma unroll
  for (int i = 0; i < 8; ++i) { r[i] = lo[i]; r[i + 8] = hi[i]; }
  return r;
}

// ---------------------------------------------------------------------------
// Utility kernels
// ---------------------------------------------------------------------------
__global__ void zero_f16_kernel(_Float16* p, size_t n) {
  size_t i = (size_t)blockIdx.x * blockDim.x + threadIdx.x;
  size_t st = (size_t)gridDim.x * blockDim.x;
  for (; i < n; i += st) p[i] = (_Float16)0.0f;
}

__global__ void f32_to_f16_kernel(const float* __restrict__ s, _Float16* __restrict__ d, int n) {
  int i = blockIdx.x * blockDim.x + threadIdx.x;
  if (i < n) d[i] = (_Float16)s[i];
}

__global__ void pad_rows_f16_kernel(const float* __restrict__ s, _Float16* __restrict__ d,
                                    int N, int Npad, int K) {
  int t = blockIdx.x * blockDim.x + threadIdx.x;
  if (t >= Npad * K) return;
  int n = t / K, k = t % K;
  d[t] = (n < N) ? (_Float16)s[n * K + k] : (_Float16)0.0f;
}

__global__ void pad_vec_f32_kernel(const float* __restrict__ s, float* __restrict__ d,
                                   int N, int Npad) {
  int t = blockIdx.x * blockDim.x + threadIdx.x;
  if (t >= Npad) return;
  d[t] = (t < N) ? s[t] : 0.0f;
}

// conv_w (3,48,64,9) f32 -> (3,9,48,64) f16
__global__ void conv_w_reorder_kernel(const float* __restrict__ s, _Float16* __restrict__ d) {
  int t = blockIdx.x * blockDim.x + threadIdx.x;
  if (t >= 3 * 9 * MIDn * Cn) return;
  int c = t & 63;
  int o = (t >> 6) % MIDn;
  int k = (t / (Cn * MIDn)) % 9;
  int i = t / (Cn * MIDn * 9);
  d[t] = (_Float16)s[(((size_t)i * MIDn + o) * Cn + c) * 9 + k];
}

// cat_w (64, nin*48, 9) f32 -> (nin*9, 64, 64) f16, cin padded 48->64 with 0
__global__ void cat_w_reorder_kernel(const float* __restrict__ s, _Float16* __restrict__ d,
                                     int nin) {
  int t = blockIdx.x * blockDim.x + threadIdx.x;
  if (t >= nin * 9 * Cn * Cn) return;
  int cin = t & 63;
  int o   = (t >> 6) & 63;
  int k   = (t / (Cn * Cn)) % 9;
  int ti  = t / (Cn * Cn * 9);
  float v = 0.0f;
  if (cin < MIDn) v = s[((size_t)o * (nin * MIDn) + (ti * MIDn + cin)) * 9 + k];
  d[((size_t)(ti * 9 + k) * Cn + o) * Cn + cin] = (_Float16)v;
}

__global__ void anchors_init_kernel(const float* __restrict__ prior_emb, float* __restrict__ anch) {
  int t = blockIdx.x * blockDim.x + threadIdx.x;  // (b*192+p)*3+k
  if (t >= MROWS * 3) return;
  int k = t % 3;
  int p = (t / 3) % NPp;
  anch[t] = prior_emb[p * 3 + k];
}

__global__ void priors0_kernel(const float* __restrict__ prior_emb, float* __restrict__ priors) {
  int t = blockIdx.x * blockDim.x + threadIdx.x;  // (b*192+p)*36+s
  if (t >= MROWS * NSn) return;
  int s = t % NSn;
  int p = (t / NSn) % NPp;
  int j = (int)(((double)s / 35.0) * 71.0);       // SAMPLE_X_IDX[s]
  float sy = prior_emb[p * 3 + 0];
  float sx = prior_emb[p * 3 + 1];
  float th = prior_emb[p * 3 + 2];
  float prior_y = 1.0f - (float)j * (1.0f / 71.0f);
  priors[t] = sx + (1.0f - prior_y - sy) * 320.0f / tanf(th * PI_F) * (1.0f / 799.0f);
}

// ---------------------------------------------------------------------------
// Bilinear prior pooling -> pooled_t (MROWS, SPAD, 64) f16, s offset +4, pads zero
// ---------------------------------------------------------------------------
__global__ void pool_kernel(const float* __restrict__ feat, int H, int W,
                            const float* __restrict__ priors, _Float16* __restrict__ pooled) {
  int t = blockIdx.x * blockDim.x + threadIdx.x;  // row*36+s
  if (t >= MROWS * NSn) return;
  int s   = t % NSn;
  int row = t / NSn;
  int b   = row / NPp;
  float x = priors[t] * (float)(W - 1);
  int jidx = (int)(((double)(35 - s) / 35.0) * 71.0);  // PRIOR_FEAT_YS
  float y = ((float)jidx * (1.0f / 71.0f)) * (float)(H - 1);
  float x0f = floorf(x), y0f = floorf(y);
  int x0 = (int)x0f, y0 = (int)y0f;
  float wx1 = x - x0f, wy1 = y - y0f;
  bool vx0 = (x0 >= 0) && (x0 <= W - 1);
  bool vx1 = (x0 + 1 >= 0) && (x0 + 1 <= W - 1);
  bool vy0 = (y0 >= 0) && (y0 <= H - 1);
  bool vy1 = (y0 + 1 >= 0) && (y0 + 1 <= H - 1);
  int cx0 = min(max(x0, 0), W - 1), cx1 = min(max(x0 + 1, 0), W - 1);
  int cy0 = min(max(y0, 0), H - 1), cy1 = min(max(y0 + 1, 0), H - 1);
  float w00 = (1.0f - wx1) * (1.0f - wy1) * ((vx0 && vy0) ? 1.0f : 0.0f);
  float w10 = wx1 * (1.0f - wy1) * ((vx1 && vy0) ? 1.0f : 0.0f);
  float w01 = (1.0f - wx1) * wy1 * ((vx0 && vy1) ? 1.0f : 0.0f);
  float w11 = wx1 * wy1 * ((vx1 && vy1) ? 1.0f : 0.0f);
  const float* fb = feat + (size_t)b * Cn * H * W;
  _Float16* op = pooled + (size_t)row * (SPAD * Cn) + (size_t)(s + 4) * Cn;
  int i00 = cy0 * W + cx0, i10 = cy0 * W + cx1, i01 = cy1 * W + cx0, i11 = cy1 * W + cx1;
  for (int c = 0; c < Cn; ++c) {
    const float* fc = fb + (size_t)c * H * W;
    float v = fc[i00] * w00 + fc[i10] * w10 + fc[i01] * w01 + fc[i11] * w11;
    op[c] = (_Float16)v;
  }
}

// ---------------------------------------------------------------------------
// Conv9 as 9 shifted tap-GEMMs via WMMA. Input layout (MROWS, SPAD, 64) f16.
// W layout: (nin*9, N, 64) f16 (B^T per tap).  blockIdx.z = output position s.
// out_mode 0: feats layout (m, SPAD, 64) at [(m)*SPAD*64 + (s+4)*64 + n]
// out_mode 1: A_fc layout [m*2304 + n*36 + s]
// Epilogue: relu(acc * scale[n] + bias[n]) -> f16
// ---------------------------------------------------------------------------
__global__ void __launch_bounds__(256)
conv_gemm_kernel(const _Float16* __restrict__ A0, const _Float16* __restrict__ A1,
                 const _Float16* __restrict__ A2, int nin,
                 const _Float16* __restrict__ W,
                 const float* __restrict__ scale, const float* __restrict__ bias,
                 _Float16* __restrict__ Out, int N, int out_mode) {
  int lane = threadIdx.x & 31;
  int wv   = threadIdx.x >> 5;
  int tiles_m = MROWS >> 4;
  int tiles_n = N >> 4;
  int tile = blockIdx.x * 8 + wv;
  if (tile >= tiles_m * tiles_n) return;
  int mt = tile % tiles_m, nt = tile / tiles_m;
  int z = blockIdx.z;  // s in 0..35
  f8 acc = {};
  for (int t = 0; t < nin; ++t) {
    const _Float16* Ai = (t == 0) ? A0 : ((t == 1) ? A1 : A2);
    const _Float16* Abase = Ai + (size_t)(mt * 16) * (SPAD * Cn) + (size_t)z * Cn;
#pragma unroll
    for (int k = 0; k < 9; ++k) {
      const _Float16* Ak = Abase + k * Cn;
      const _Float16* Wk = W + (size_t)((t * 9 + k) * N + nt * 16) * Cn;
      h16 af0 = frag_a16(Ak, SPAD * Cn, lane);
      h16 bf0 = frag_b16(Wk, Cn, lane);
      acc = WMMA_F32_F16(af0, bf0, acc);
      h16 af1 = frag_a16(Ak + 32, SPAD * Cn, lane);
      h16 bf1 = frag_b16(Wk + 32, Cn, lane);
      acc = WMMA_F32_F16(af1, bf1, acc);
    }
  }
  int n = nt * 16 + (lane & 15);
  int mbase = mt * 16 + ((lane >> 4) << 3);
  float sc = scale[n], bi = bias[n];
  if (out_mode == 0) {
#pragma unroll
    for (int v = 0; v < 8; ++v)
      Out[(size_t)(mbase + v) * (SPAD * Cn) + (size_t)(z + 4) * Cn + n] =
          (_Float16)fmaxf(acc[v] * sc + bi, 0.0f);
  } else {
#pragma unroll
    for (int v = 0; v < 8; ++v)
      Out[(size_t)(mbase + v) * KFC + n * NSn + z] =
          (_Float16)fmaxf(acc[v] * sc + bi, 0.0f);
  }
}

// ---------------------------------------------------------------------------
// LDS-staged fc GEMM: (6144 x 2304) @ Bt(64 x 2304)^T + bias -> f32 (6144 x 64)
// Block = 8 waves; block tile 128 x 64; K streamed in KC=64 panels, double-
// buffered via global_load_async_to_lds_b128 (ASYNCcnt) so DMA overlaps WMMA.
// ---------------------------------------------------------------------------
__global__ void __launch_bounds__(256)
gemm_fc_lds_kernel(const _Float16* __restrict__ A,   // (MROWS, KFC)
                   const _Float16* __restrict__ Bt,  // (Cn, KFC)
                   const float* __restrict__ bias,
                   float* __restrict__ Out) {        // (MROWS, Cn)
  __shared__ __align__(16) _Float16 smA[2][128 * KC];  // 2 x 16 KB
  __shared__ __align__(16) _Float16 smB[2][Cn * KC];   // 2 x  8 KB
  int tid  = threadIdx.x;
  int lane = tid & 31;
  int wv   = tid >> 5;
  int m0   = blockIdx.x * 128;

  auto load_panel = [&](int buf, int kbase) {
    // A panel: 128 rows x 64 halfs = 1024 x 16B, 4 per thread
#pragma unroll
    for (int i = 0; i < 4; ++i) {
      int q  = tid + i * 256;
      int r  = q >> 3;
      int c8 = (q & 7) << 3;
      async_ld_b128((uint32_t)(uintptr_t)&smA[buf][q << 3],
                    A + (size_t)(m0 + r) * KFC + kbase + c8);
    }
    // B panel: 64 rows x 64 halfs = 512 x 16B, 2 per thread
#pragma unroll
    for (int i = 0; i < 2; ++i) {
      int q  = tid + i * 256;
      int r  = q >> 3;
      int c8 = (q & 7) << 3;
      async_ld_b128((uint32_t)(uintptr_t)&smB[buf][q << 3],
                    Bt + (size_t)r * KFC + kbase + c8);
    }
  };

  f8 acc[4];
#pragma unroll
  for (int nt = 0; nt < 4; ++nt) acc[nt] = (f8){};

  load_panel(0, 0);
  wait_async0();
  __syncthreads();

  const int NPAN = KFC / KC;  // 36
  for (int c = 0; c < NPAN; ++c) {
    int buf = c & 1;
    if (c + 1 < NPAN) load_panel(buf ^ 1, (c + 1) * KC);
    const _Float16* Abuf = &smA[buf][wv * 16 * KC];
    const _Float16* Bbuf = &smB[buf][0];
#pragma unroll
    for (int kt = 0; kt < KC; kt += 32) {
      h16 af = frag_a16(Abuf + kt, KC, lane);
#pragma unroll
      for (int nt = 0; nt < 4; ++nt) {
        h16 bf = frag_b16(Bbuf + nt * 16 * KC + kt, KC, lane);
        acc[nt] = WMMA_F32_F16(af, bf, acc[nt]);
      }
    }
    wait_async0();      // prefetch issued before compute -> DMA overlapped
    __syncthreads();
  }

  int mbase = m0 + wv * 16 + ((lane >> 4) << 3);
  int nl    = lane & 15;
#pragma unroll
  for (int nt = 0; nt < 4; ++nt) {
    int n = nt * 16 + nl;
    float bi = bias[n];
#pragma unroll
    for (int v = 0; v < 8; ++v)
      Out[(size_t)(mbase + v) * Cn + n] = acc[nt][v] + bi;
  }
}

// ---------------------------------------------------------------------------
// Generic batched WMMA GEMM: C[b] = A[b] (MxK, row-major) @ B[b]^T (B given NxK)
// K-loop software-pipelined (next fragments in flight during current WMMA).
// epi 0: f32 out = acc + bias[n]
// epi 1: f16 out = relu(acc*scale[n] + bias[n])
// epi 2: f16 out transposed: out[n*ldo + m] = acc + bias[n]
// epi 3: f32 out = acc * alpha
// ---------------------------------------------------------------------------
__global__ void __launch_bounds__(256)
gemm_kernel(const _Float16* __restrict__ A, size_t sAb, int lda,
            const _Float16* __restrict__ Bt, size_t sBb, int ldb,
            void* __restrict__ Out, size_t sOb, int ldo,
            int M, int N, int K,
            const float* __restrict__ scale, const float* __restrict__ bias,
            float alpha, int epi) {
  int lane = threadIdx.x & 31;
  int wv   = threadIdx.x >> 5;
  int tiles_m = M >> 4, tiles_n = N >> 4;
  int tile = blockIdx.x * 8 + wv;
  if (tile >= tiles_m * tiles_n) return;
  int mt = tile % tiles_m, nt = tile / tiles_m;
  int b  = blockIdx.z;
  const _Float16* Ab = A + (size_t)b * sAb + (size_t)(mt * 16) * lda;
  const _Float16* Bb = Bt + (size_t)b * sBb + (size_t)(nt * 16) * ldb;
  f8 acc = {};
  h16 a0 = frag_a16(Ab, lda, lane);
  h16 b0 = frag_b16(Bb, ldb, lane);
  for (int k = 0; k < K; k += 32) {
    h16 a1 = a0, b1 = b0;
    if (k + 32 < K) {
      a1 = frag_a16(Ab + k + 32, lda, lane);
      b1 = frag_b16(Bb + k + 32, ldb, lane);
    }
    acc = WMMA_F32_F16(a0, b0, acc);
    a0 = a1;
    b0 = b1;
  }
  int n = nt * 16 + (lane & 15);
  int mbase = mt * 16 + ((lane >> 4) << 3);
  float sc = scale ? scale[n] : 1.0f;
  float bi = bias ? bias[n] : 0.0f;
  if (epi == 0) {
    float* o = (float*)Out + (size_t)b * sOb;
#pragma unroll
    for (int v = 0; v < 8; ++v) o[(size_t)(mbase + v) * ldo + n] = acc[v] + bi;
  } else if (epi == 1) {
    _Float16* o = (_Float16*)Out + (size_t)b * sOb;
#pragma unroll
    for (int v = 0; v < 8; ++v)
      o[(size_t)(mbase + v) * ldo + n] = (_Float16)fmaxf(acc[v] * sc + bi, 0.0f);
  } else if (epi == 2) {
    _Float16* o = (_Float16*)Out + (size_t)b * sOb;
#pragma unroll
    for (int v = 0; v < 8; ++v)
      o[(size_t)n * ldo + (mbase + v)] = (_Float16)(acc[v] + bi);
  } else {
    float* o = (float*)Out + (size_t)b * sOb;
#pragma unroll
    for (int v = 0; v < 8; ++v) o[(size_t)(mbase + v) * ldo + n] = acc[v] * alpha;
  }
}

// ---------------------------------------------------------------------------
// LayerNorm(64) + relu -> roi (f32), and query = relu(roi*fq_w[p]+fq_b[p]) f16
// ---------------------------------------------------------------------------
__global__ void ln_kernel(const float* __restrict__ x, const float* __restrict__ ln_g,
                          const float* __restrict__ ln_b, const float* __restrict__ fq_w,
                          const float* __restrict__ fq_b, float* __restrict__ roi,
                          _Float16* __restrict__ query) {
  int lane = threadIdx.x & 31;
  int wave = threadIdx.x >> 5;
  int row  = blockIdx.x * 8 + wave;
  if (row >= MROWS) return;
  const float* xr = x + (size_t)row * Cn;
  float a0 = xr[lane], a1 = xr[lane + 32];
  float s1 = a0 + a1, s2 = a0 * a0 + a1 * a1;
#pragma unroll
  for (int off = 16; off > 0; off >>= 1) {
    s1 += __shfl_xor(s1, off, 32);
    s2 += __shfl_xor(s2, off, 32);
  }
  float mu  = s1 * (1.0f / 64.0f);
  float var = s2 * (1.0f / 64.0f) - mu * mu;
  float inv = rsqrtf(var + 1e-5f);
  int p = row % NPp;
  float fw = fq_w[p], fb = fq_b[p];
  for (int c = lane; c < Cn; c += 32) {
    float y = (xr[c] - mu) * inv * ln_g[c] + ln_b[c];
    y = fmaxf(y, 0.0f);
    roi[(size_t)row * Cn + c] = y;
    query[(size_t)row * Cn + c] = (_Float16)fmaxf(y * fw + fb, 0.0f);
  }
}

// Gather 250 resize_flatten positions into G (B, 256, 64) f16, pad rows zero
__global__ void gather_kernel(const float* __restrict__ feat, int H, int W,
                              _Float16* __restrict__ G) {
  int t = blockIdx.x * blockDim.x + threadIdx.x;  // b*256*64 total
  if (t >= Bn * 256 * Cn) return;
  int c = t & 63;
  int n = (t >> 6) & 255;
  int b = t >> 14;
  _Float16 v = (_Float16)0.0f;
  if (n < 250) {
    int i = n / 25, j = n % 25;
    int sr = H / 10, sc = W / 25;
    v = (_Float16)feat[(((size_t)b * Cn + c) * H + i * sr) * W + j * sc];
  }
  G[t] = v;
}

// Row softmax over 250 valid cols of (6144, 256) f32 -> f16, pad zero
__global__ void softmax_kernel(const float* __restrict__ sim, _Float16* __restrict__ simh) {
  int lane = threadIdx.x & 31;
  int wave = threadIdx.x >> 5;
  int row  = blockIdx.x * 8 + wave;
  if (row >= MROWS) return;
  const float* s = sim + (size_t)row * 256;
  float m = -1e30f;
  for (int n = lane; n < 250; n += 32) m = fmaxf(m, s[n]);
#pragma unroll
  for (int off = 16; off > 0; off >>= 1) m = fmaxf(m, __shfl_xor(m, off, 32));
  float e[8];
  int cnt = 0;
  float sum = 0.0f;
  for (int n = lane; n < 250; n += 32) {
    float v = __expf(s[n] - m);
    e[cnt++] = v;
    sum += v;
  }
#pragma unroll
  for (int off = 16; off > 0; off >>= 1) sum += __shfl_xor(sum, off, 32);
  float inv = 1.0f / sum;
  _Float16* o = simh + (size_t)row * 256;
  cnt = 0;
  for (int n = lane; n < 250; n += 32) o[n] = (_Float16)(e[cnt++] * inv);
  for (int n = 250 + lane; n < 256; n += 32) o[n] = (_Float16)0.0f;
}

// fc_feat = roi + relu(ctx*aw_w[p] + aw_b[p]) -> f16
__global__ void combine_kernel(const float* __restrict__ roi, const float* __restrict__ ctx,
                               const float* __restrict__ aw_w, const float* __restrict__ aw_b,
                               _Float16* __restrict__ fc16) {
  int t = blockIdx.x * blockDim.x + threadIdx.x;
  if (t >= MROWS * Cn) return;
  int p = (t >> 6) % NPp;
  float c = fmaxf(ctx[t] * aw_w[p] + aw_b[p], 0.0f);
  fc16[t] = (_Float16)(roi[t] + c);
}

// Stage finalize: update anchors, build output slice, produce next priors
__global__ void finalize_kernel(const float* __restrict__ clsf, const float* __restrict__ regf,
                                float* __restrict__ anch, float* __restrict__ priors,
                                float* __restrict__ out, int stage, int write_priors) {
  int t = blockIdx.x * blockDim.x + threadIdx.x;  // row = b*192+p
  if (t >= MROWS) return;
  const float* r = regf + (size_t)t * 80;
  float sy = anch[t * 3 + 0] + r[0];
  float sx = anch[t * 3 + 1] + r[1];
  float th = anch[t * 3 + 2] + r[2];
  anch[t * 3 + 0] = sy;
  anch[t * 3 + 1] = sx;
  anch[t * 3 + 2] = th;
  float* o = out + ((size_t)stage * MROWS + t) * 78;
  o[0] = clsf[t * 16 + 0];
  o[1] = clsf[t * 16 + 1];
  o[2] = sy; o[3] = sx; o[4] = th;
  o[5] = r[3];
  float inv_tan = 1.0f / tanf(th * PI_F);
  float ux[NOn];
  for (int j = 0; j < NOn; ++j) {
    float prior_y = 1.0f - (float)j * (1.0f / 71.0f);
    float u = sx + (1.0f - prior_y - sy) * 320.0f * inv_tan * (1.0f / 799.0f);
    ux[j] = u;
    o[6 + j] = u + r[4 + j];
  }
  if (write_priors) {
    for (int s = 0; s < NSn; ++s) {
      int j = (int)(((double)s / 35.0) * 71.0);
      priors[(size_t)t * NSn + s] = ux[j];
    }
  }
}

// ---------------------------------------------------------------------------
// Host orchestration
// ---------------------------------------------------------------------------
static inline int cdiv(int a, int b) { return (a + b - 1) / b; }

extern "C" void kernel_launch(void* const* d_in, const int* in_sizes, int n_in,
                              void* d_out, int out_size, void* d_ws, size_t ws_size,
                              hipStream_t stream) {
  (void)in_sizes; (void)n_in; (void)out_size; (void)ws_size;
  const float* x0        = (const float*)d_in[0];
  const float* x1        = (const float*)d_in[1];
  const float* x2        = (const float*)d_in[2];
  const float* prior_emb = (const float*)d_in[3];
  const float* cls_fc_w  = (const float*)d_in[4];
  const float* cls_fc_b  = (const float*)d_in[5];
  const float* reg_fc_w  = (const float*)d_in[6];
  const float* reg_fc_b  = (const float*)d_in[7];
  const float* cls_out_w = (const float*)d_in[8];
  const float* cls_out_b = (const float*)d_in[9];
  const float* reg_out_w = (const float*)d_in[10];
  const float* reg_out_b = (const float*)d_in[11];
  const float* conv_w    = (const float*)d_in[12];
  const float* conv_s    = (const float*)d_in[13];
  const float* conv_b    = (const float*)d_in[14];
  const float* cat_w0    = (const float*)d_in[15];
  const float* cat_w1    = (const float*)d_in[16];
  const float* cat_w2    = (const float*)d_in[17];
  const float* cat_s     = (const float*)d_in[18];
  const float* cat_b     = (const float*)d_in[19];
  const float* fc_w      = (const float*)d_in[20];
  const float* fc_b      = (const float*)d_in[21];
  const float* ln_g      = (const float*)d_in[22];
  const float* ln_b      = (const float*)d_in[23];
  const float* fk_w      = (const float*)d_in[24];
  const float* fk_s      = (const float*)d_in[25];
  const float* fk_b      = (const float*)d_in[26];
  const float* fv_w      = (const float*)d_in[27];
  const float* fv_bias   = (const float*)d_in[28];
  const float* fq_w      = (const float*)d_in[29];
  const float* fq_b      = (const float*)d_in[30];
  const float* aw_w      = (const float*)d_in[31];
  const float* aw_b      = (const float*)d_in[32];

  char* ws = (char*)d_ws;
  size_t off = 0;
  auto alloc = [&](size_t bytes) -> char* {
    size_t a = (off + 255) & ~(size_t)255;
    off = a + bytes;
    return ws + a;
  };

  const size_t POOL_ELEMS = (size_t)MROWS * SPAD * Cn;
  _Float16* pooled = (_Float16*)alloc(POOL_ELEMS * 2);
  _Float16* feats0 = (_Float16*)alloc(POOL_ELEMS * 2);
  _Float16* feats1 = (_Float16*)alloc(POOL_ELEMS * 2);
  _Float16* feats2 = (_Float16*)alloc(POOL_ELEMS * 2);
  _Float16* Afc    = (_Float16*)alloc((size_t)MROWS * KFC * 2);
  float*    roipre = (float*)alloc((size_t)MROWS * Cn * 4);
  float*    roi    = (float*)alloc((size_t)MROWS * Cn * 4);
  _Float16* query  = (_Float16*)alloc((size_t)MROWS * Cn * 2);
  _Float16* G      = (_Float16*)alloc((size_t)Bn * 256 * Cn * 2);
  _Float16* Kb     = (_Float16*)alloc((size_t)Bn * 256 * Cn * 2);
  _Float16* Vt     = (_Float16*)alloc((size_t)Bn * Cn * 256 * 2);
  float*    sim    = (float*)alloc((size_t)Bn * NPp * 256 * 4);
  _Float16* simh   = (_Float16*)alloc((size_t)Bn * NPp * 256 * 2);
  float*    ctx    = (float*)alloc((size_t)MROWS * Cn * 4);
  _Float16* fc16   = (_Float16*)alloc((size_t)MROWS * Cn * 2);
  _Float16* h1     = (_Float16*)alloc((size_t)MROWS * Cn * 2);
  _Float16* h2     = (_Float16*)alloc((size_t)MROWS * Cn * 2);
  float*    clsf   = (float*)alloc((size_t)MROWS * 16 * 4);
  float*    regf   = (float*)alloc((size_t)MROWS * 80 * 4);
  float*    anch   = (float*)alloc((size_t)MROWS * 3 * 4);
  float*    priors = (float*)alloc((size_t)MROWS * NSn * 4);
  // f16 weights
  _Float16* fcw16   = (_Float16*)alloc((size_t)Cn * KFC * 2);
  _Float16* clsfcw  = (_Float16*)alloc((size_t)2 * Cn * Cn * 2);
  _Float16* regfcw  = (_Float16*)alloc((size_t)2 * Cn * Cn * 2);
  _Float16* clsow   = (_Float16*)alloc((size_t)16 * Cn * 2);
  _Float16* regow   = (_Float16*)alloc((size_t)80 * Cn * 2);
  float*    clsob   = (float*)alloc(16 * 4);
  float*    regob   = (float*)alloc(80 * 4);
  _Float16* fkw16   = (_Float16*)alloc((size_t)Cn * Cn * 2);
  _Float16* fvw16   = (_Float16*)alloc((size_t)Cn * Cn * 2);
  _Float16* convwt  = (_Float16*)alloc((size_t)3 * 9 * MIDn * Cn * 2);
  _Float16* catwt0  = (_Float16*)alloc((size_t)9 * Cn * Cn * 2);
  _Float16* catwt1  = (_Float16*)alloc((size_t)18 * Cn * Cn * 2);
  _Float16* catwt2  = (_Float16*)alloc((size_t)27 * Cn * Cn * 2);

  // ---- one-time (per launch) prep ----
  zero_f16_kernel<<<2048, 256, 0, stream>>>(pooled, POOL_ELEMS);
  zero_f16_kernel<<<2048, 256, 0, stream>>>(feats0, POOL_ELEMS);
  zero_f16_kernel<<<2048, 256, 0, stream>>>(feats1, POOL_ELEMS);
  zero_f16_kernel<<<2048, 256, 0, stream>>>(feats2, POOL_ELEMS);

  f32_to_f16_kernel<<<cdiv(Cn * KFC, 256), 256, 0, stream>>>(fc_w, fcw16, Cn * KFC);
  f32_to_f16_kernel<<<cdiv(2 * Cn * Cn, 256), 256, 0, stream>>>(cls_fc_w, clsfcw, 2 * Cn * Cn);
  f32_to_f16_kernel<<<cdiv(2 * Cn * Cn, 256), 256, 0, stream>>>(reg_fc_w, regfcw, 2 * Cn * Cn);
  f32_to_f16_kernel<<<cdiv(Cn * Cn, 256), 256, 0, stream>>>(fk_w, fkw16, Cn * Cn);
  f32_to_f16_kernel<<<cdiv(Cn * Cn, 256), 256, 0, stream>>>(fv_w, fvw16, Cn * Cn);
  pad_rows_f16_kernel<<<cdiv(16 * Cn, 256), 256, 0, stream>>>(cls_out_w, clsow, 2, 16, Cn);
  pad_rows_f16_kernel<<<cdiv(80 * Cn, 256), 256, 0, stream>>>(reg_out_w, regow, 76, 80, Cn);
  pad_vec_f32_kernel<<<1, 256, 0, stream>>>(cls_out_b, clsob, 2, 16);
  pad_vec_f32_kernel<<<1, 256, 0, stream>>>(reg_out_b, regob, 76, 80);
  conv_w_reorder_kernel<<<cdiv(3 * 9 * MIDn * Cn, 256), 256, 0, stream>>>(conv_w, convwt);
  cat_w_reorder_kernel<<<cdiv(9 * Cn * Cn, 256), 256, 0, stream>>>(cat_w0, catwt0, 1);
  cat_w_reorder_kernel<<<cdiv(18 * Cn * Cn, 256), 256, 0, stream>>>(cat_w1, catwt1, 2);
  cat_w_reorder_kernel<<<cdiv(27 * Cn * Cn, 256), 256, 0, stream>>>(cat_w2, catwt2, 3);
  anchors_init_kernel<<<cdiv(MROWS * 3, 256), 256, 0, stream>>>(prior_emb, anch);
  priors0_kernel<<<cdiv(MROWS * NSn, 256), 256, 0, stream>>>(prior_emb, priors);

  const float* pyr[3] = {x2, x1, x0};
  const int Hs[3] = {10, 20, 40};
  const int Wsz[3] = {25, 50, 100};
  _Float16* featsArr[3] = {feats0, feats1, feats2};
  const _Float16* catw[3] = {catwt0, catwt1, catwt2};

  for (int s = 0; s < 3; ++s) {
    // pooling (stage s priors, pyramid level s)
    pool_kernel<<<cdiv(MROWS * NSn, 256), 256, 0, stream>>>(pyr[s], Hs[s], Wsz[s], priors, pooled);

    // feats[s] = relu(conv9(pooled, conv_w[s]) * conv_s[s] + conv_b[s])  (WMMA taps)
    conv_gemm_kernel<<<dim3(cdiv((MROWS / 16) * (MIDn / 16), 8), 1, NSn), 256, 0, stream>>>(
        pooled, nullptr, nullptr, 1, convwt + (size_t)s * 9 * MIDn * Cn,
        conv_s + s * MIDn, conv_b + s * MIDn, featsArr[s], MIDn, 0);

    // roi conv: cat over stages -> A_fc (flattened c*36+s) with relu(scale,bias)
    conv_gemm_kernel<<<dim3(cdiv((MROWS / 16) * (Cn / 16), 8), 1, NSn), 256, 0, stream>>>(
        feats0, feats1, feats2, s + 1, catw[s], cat_s + s * Cn, cat_b + s * Cn, Afc, Cn, 1);

    // fc: (6144 x 2304) @ fc_w^T + fc_b  (async-LDS double-buffered WMMA)
    gemm_fc_lds_kernel<<<MROWS / 128, 256, 0, stream>>>(Afc, fcw16, fc_b, roipre);

    // LN + relu, query
    ln_kernel<<<MROWS / 8, 256, 0, stream>>>(roipre, ln_g, ln_b, fq_w, fq_b, roi, query);

    // attention operands at 250 sampled pixels
    gather_kernel<<<cdiv(Bn * 256 * Cn, 256), 256, 0, stream>>>(pyr[s], Hs[s], Wsz[s], G);
    // key = relu(G @ fk_w^T * fk_s + fk_b)
    gemm_kernel<<<dim3(cdiv(16 * 4, 8), 1, Bn), 256, 0, stream>>>(
        G, (size_t)256 * Cn, Cn, fkw16, 0, Cn, Kb, (size_t)256 * Cn, Cn,
        256, Cn, Cn, fk_s, fk_b, 1.0f, 1);
    // value = G @ fv_w^T + fv_bias, stored transposed (64 x 256)
    gemm_kernel<<<dim3(cdiv(16 * 4, 8), 1, Bn), 256, 0, stream>>>(
        G, (size_t)256 * Cn, Cn, fvw16, 0, Cn, Vt, (size_t)Cn * 256, 256,
        256, Cn, Cn, nullptr, fv_bias, 1.0f, 2);
    // sim = (query @ key^T) * C^-0.5
    gemm_kernel<<<dim3(cdiv(12 * 16, 8), 1, Bn), 256, 0, stream>>>(
        query, (size_t)NPp * Cn, Cn, Kb, (size_t)256 * Cn, Cn, sim, (size_t)NPp * 256, 256,
        NPp, 256, Cn, nullptr, nullptr, 0.125f, 3);
    softmax_kernel<<<MROWS / 8, 256, 0, stream>>>(sim, simh);
    // ctx = softmax(sim) @ value
    gemm_kernel<<<dim3(cdiv(12 * 4, 8), 1, Bn), 256, 0, stream>>>(
        simh, (size_t)NPp * 256, 256, Vt, (size_t)Cn * 256, 256, ctx, (size_t)NPp * Cn, Cn,
        NPp, Cn, 256, nullptr, nullptr, 1.0f, 0);
    combine_kernel<<<cdiv(MROWS * Cn, 256), 256, 0, stream>>>(roi, ctx, aw_w, aw_b, fc16);

    // cls head
    gemm_kernel<<<dim3(cdiv(384 * 4, 8), 1, 1), 256, 0, stream>>>(
        fc16, 0, Cn, clsfcw, 0, Cn, h1, 0, Cn, MROWS, Cn, Cn, nullptr, cls_fc_b, 1.0f, 1);
    gemm_kernel<<<dim3(cdiv(384 * 4, 8), 1, 1), 256, 0, stream>>>(
        h1, 0, Cn, clsfcw + Cn * Cn, 0, Cn, h2, 0, Cn, MROWS, Cn, Cn, nullptr,
        cls_fc_b + Cn, 1.0f, 1);
    gemm_kernel<<<dim3(cdiv(384 * 1, 8), 1, 1), 256, 0, stream>>>(
        h2, 0, Cn, clsow, 0, Cn, clsf, 0, 16, MROWS, 16, Cn, nullptr, clsob, 1.0f, 0);
    // reg head
    gemm_kernel<<<dim3(cdiv(384 * 4, 8), 1, 1), 256, 0, stream>>>(
        fc16, 0, Cn, regfcw, 0, Cn, h1, 0, Cn, MROWS, Cn, Cn, nullptr, reg_fc_b, 1.0f, 1);
    gemm_kernel<<<dim3(cdiv(384 * 4, 8), 1, 1), 256, 0, stream>>>(
        h1, 0, Cn, regfcw + Cn * Cn, 0, Cn, h2, 0, Cn, MROWS, Cn, Cn, nullptr,
        reg_fc_b + Cn, 1.0f, 1);
    gemm_kernel<<<dim3(cdiv(384 * 5, 8), 1, 1), 256, 0, stream>>>(
        h2, 0, Cn, regow, 0, Cn, regf, 0, 80, MROWS, 80, Cn, nullptr, regob, 1.0f, 0);

    finalize_kernel<<<cdiv(MROWS, 256), 256, 0, stream>>>(
        clsf, regf, anch, priors, (float*)d_out, s, (s < 2) ? 1 : 0);
  }
}
